// _SigLIPKMoeHead_16724602650680
// MI455X (gfx1250) — matrix operand order repaired and checked by reading, not compiled, not run on hardware
//
#include <hip/hip_runtime.h>
#include <hip/hip_bf16.h>

// ---------------------------------------------------------------------------
// Sizes (compile-time constants from the reference)
// ---------------------------------------------------------------------------
#define BB   64
#define SS   1024
#define DD   1152
#define TT   4
#define EE   8
#define FF_  4304
#define NHH  16
#define DHH  72     // DD / NHH

typedef __attribute__((ext_vector_type(16))) __bf16 v16bf;
typedef __attribute__((ext_vector_type(8)))  float  v8f;

struct __attribute__((aligned(16))) F4 { float x, y, z, w; };
struct __attribute__((aligned(16))) U4 { unsigned int x, y, z, w; };
struct __attribute__((aligned(8)))  U2 { unsigned int x, y; };

__device__ __forceinline__ unsigned short f2bf(float f) {
    unsigned int u = __float_as_uint(f);
    u = (u + 0x7FFFu + ((u >> 16) & 1u)) >> 16;
    return (unsigned short)u;
}
__device__ __forceinline__ float bf2f(unsigned short h) {
    return __uint_as_float(((unsigned int)h) << 16);
}
// packed fp32x2 -> bf16x2 (RNE) in one VALU op
__device__ __forceinline__ unsigned int pk_bf16(float a, float b) {
    unsigned int r;
    asm volatile("v_cvt_pk_bf16_f32 %0, %1, %2" : "=v"(r) : "v"(a), "v"(b));
    return r;
}
// async global->LDS 16B copy (per-lane), tracked by ASYNCcnt
__device__ __forceinline__ void async_ld_b128(unsigned lds_off, unsigned long long gaddr) {
    asm volatile("global_load_async_to_lds_b128 %0, %1, off"
                 :: "v"(lds_off), "v"(gaddr) : "memory");
}
__device__ __forceinline__ void wait_async0() {
    asm volatile("s_wait_asynccnt 0" ::: "memory");
}
__device__ __forceinline__ unsigned lds_addr32(const void* p) {
    return (unsigned)(size_t)p;   // generic LDS pointer: low 32 bits = LDS offset
}

// ---------------------------------------------------------------------------
// Fragment loads from LDS, per CDNA5 WMMA VGPR layouts (CDNA5 ISA)
// A 16x32 bf16: lane m (0..15 dup); lanes<16 K {0..7,16..23}, lanes>=16
//               K {8..15,24..31}.  B 32x16 bf16: lane n; same K order as A.
//               Tiles row-major [row][k], stride 40 halves.
// ---------------------------------------------------------------------------
__device__ __forceinline__ v16bf lds_frag_a(const unsigned short* As, int mOff, int lane) {
    int m    = mOff + (lane & 15);
    int koff = (lane >> 4) * 8;
    const unsigned short* p = As + m * 40 + koff;
    union { v16bf v; U4 q[2]; } u;
    u.q[0] = *(const U4*)(p);
    u.q[1] = *(const U4*)(p + 16);
    return u.v;
}
__device__ __forceinline__ v16bf lds_frag_b(const unsigned short* Bs, int nOff, int lane) {
    int n     = nOff + (lane & 15);
    int kbase = (lane >> 4) * 8;
    const unsigned short* p = Bs + n * 40 + kbase;
    union { v16bf v; U4 q[2]; } u;
    u.q[0] = *(const U4*)(p);
    u.q[1] = *(const U4*)(p + 16);
    return u.v;
}

// ---------------------------------------------------------------------------
// Generic WMMA GEMM:  C[M,N] = epilogue( A[M,K] * Bw[N,K]^T + bias[N] )
// Block tile 128x128, K-step 32, double-buffered LDS.
// 256 threads = 8 waves (4M x 2N); each wave: 2x4 v_wmma_f32_16x16x32_bf16.
// A fp32 path: load + v_cvt_pk_bf16_f32 + ds_store.
// A bf16 path: global_load_async_to_lds_b128 (ASYNCcnt).
// EPI: 0 = +bias -> bf16   1 = +bias -> f32
//      2 = gelu(x+bias)*scale -> bf16   3 = atomicAdd f32
// ---------------------------------------------------------------------------
template <typename TA, int EPI>
__global__ __launch_bounds__(256) void wmma_gemm_kernel(
    const TA* __restrict__ A, long strideAz,
    const float* __restrict__ Bw, long strideBz,
    const float* __restrict__ bias, long strideBiasz,
    float* outF, unsigned short* outH, long strideOz,
    const float* __restrict__ scale, int scaleStride, long strideScalez,
    int M, int N, int K, int lda, int ldb, int ldc)
{
    __shared__ __align__(16) unsigned short As[2][128 * 40];
    __shared__ __align__(16) unsigned short Bs[2][128 * 40];

    const int tid   = threadIdx.x;
    const int lane  = tid & 31;
    const int wave  = tid >> 5;
    const int wm    = wave >> 1;   // 0..3 -> 32-row slab
    const int wn    = wave & 1;    // 0..1 -> 64-col slab
    const int mTile = blockIdx.y * 128;
    const int nTile = blockIdx.x * 128;

    const TA*    Az    = A  + (long)blockIdx.z * strideAz;
    const float* Bz    = Bw + (long)blockIdx.z * strideBz;
    const float* biasz = bias  ? bias  + (long)blockIdx.z * strideBiasz  : nullptr;
    float*       oF    = outF  ? outF  + (long)blockIdx.z * strideOz     : nullptr;
    unsigned short* oH = outH  ? outH  + (long)blockIdx.z * strideOz     : nullptr;
    const float* sclz  = scale ? scale + (long)blockIdx.z * strideScalez : nullptr;

    auto stageA = [&](int k0, int buf) {
        if constexpr (sizeof(TA) == 4) {           // fp32 -> bf16 convert
            int row = tid >> 3;                    // 0..31
            int col = (tid & 7) * 4;               // 0..28
            for (int it = 0; it < 4; ++it) {
                int r  = row + it * 32;
                int gk = k0 + col;
                F4 v = {0.f, 0.f, 0.f, 0.f};
                if (gk < K)
                    v = *(const F4*)((const float*)Az + (long)(mTile + r) * lda + gk);
                U2 o;
                o.x = pk_bf16(v.x, v.y);
                o.y = pk_bf16(v.z, v.w);
                *(U2*)&As[buf][r * 40 + col] = o;
            }
        } else {                                   // bf16: async global->LDS
            unsigned base = lds_addr32(&As[buf][0]);
            for (int i = 0; i < 2; ++i) {
                int cid = tid + 256 * i;           // 512 segs of 16B
                int row = cid >> 2, seg = cid & 3;
                int gk  = k0 + seg * 8;
                unsigned lds = base + (unsigned)(row * 80 + seg * 16);
                if (gk < K) {
                    unsigned long long ga = (unsigned long long)(const void*)
                        ((const unsigned short*)Az + (long)(mTile + row) * lda + gk);
                    async_ld_b128(lds, ga);
                } else {
                    *(U4*)((char*)&As[buf][0] + row * 80 + seg * 16) = U4{0u, 0u, 0u, 0u};
                }
            }
        }
    };
    auto stageB = [&](int k0, int buf) {           // weights fp32 [N][K]
        int row = tid >> 3;
        int col = (tid & 7) * 4;
        for (int it = 0; it < 4; ++it) {
            int r  = row + it * 32;
            int gn = nTile + r;
            int gk = k0 + col;
            F4 v = {0.f, 0.f, 0.f, 0.f};
            if (gn < N && gk < K)
                v = *(const F4*)(Bz + (long)gn * ldb + gk);
            U2 o;
            o.x = pk_bf16(v.x, v.y);
            o.y = pk_bf16(v.z, v.w);
            *(U2*)&Bs[buf][r * 40 + col] = o;
        }
    };

    v8f acc[2][4] = {};

    stageA(0, 0);
    stageB(0, 0);
    int cur = 0;
    for (int k0 = 0; k0 < K; k0 += 32) {
        if constexpr (sizeof(TA) == 2) wait_async0();
        __syncthreads();                   // buf[cur] ready for everyone
        if (k0 + 32 < K) {                 // prefetch next tile into other buffer
            stageA(k0 + 32, cur ^ 1);
            stageB(k0 + 32, cur ^ 1);
        }
        v16bf af[2], bf[4];
        for (int i = 0; i < 2; ++i) af[i] = lds_frag_a(&As[cur][0], wm * 32 + i * 16, lane);
        for (int j = 0; j < 4; ++j) bf[j] = lds_frag_b(&Bs[cur][0], wn * 64 + j * 16, lane);
        for (int i = 0; i < 2; ++i)
            for (int j = 0; j < 4; ++j)
                acc[i][j] = __builtin_amdgcn_wmma_f32_16x16x32_bf16(
                    false, af[i], false, bf[j], (short)0, acc[i][j], false, false);
        cur ^= 1;
    }

    // ---- epilogue: lane holds N = lane&15, M = (lane>>4)*8 + r ----
    const int ln16 = lane & 15;
    const int lhi  = lane >> 4;
    for (int i = 0; i < 2; ++i) {
        int mBase = mTile + wm * 32 + i * 16 + lhi * 8;
        for (int j = 0; j < 4; ++j) {
            int n = nTile + wn * 64 + j * 16 + ln16;
            if (n < N) {
                float bb = biasz ? biasz[n] : 0.0f;
                for (int r = 0; r < 8; ++r) {
                    int   m   = mBase + r;
                    float c   = acc[i][j][r] + bb;
                    long  off = (long)m * ldc + n;
                    if constexpr (EPI == 0) {
                        oH[off] = f2bf(c);
                    } else if constexpr (EPI == 1) {
                        oF[off] = c;
                    } else if constexpr (EPI == 2) {
                        float g = 0.5f * c *
                                  (1.0f + tanhf(0.7978845608028654f * (c + 0.044715f * c * c * c)));
                        g *= sclz[(long)m * scaleStride];
                        oH[off] = f2bf(g);
                    } else {
                        atomicAdd(&oF[off], c);
                    }
                }
            }
        }
    }
}

// ---------------------------------------------------------------------------
// q = (probe @ Wq^T + bq) / sqrt(DH)   -- probe broadcast over batch, so once.
// ---------------------------------------------------------------------------
__global__ void qproj_kernel(const float* __restrict__ probe,
                             const float* __restrict__ W,
                             const float* __restrict__ bq,
                             float* __restrict__ qbuf)
{
    int idx = blockIdx.x * 256 + threadIdx.x;
    if (idx >= TT * DD) return;
    int t = idx / DD, e = idx % DD;
    const float* pr = probe + t * DD;
    const float* w  = W + (long)e * DD;
    float s = bq[e];
    for (int d = 0; d < DD; ++d) s += pr[d] * w[d];
    qbuf[idx] = s * 0.11785113019775793f;   // 1/sqrt(72)
}

// ---------------------------------------------------------------------------
// Attention: one block per (head, batch), 288 threads.
// KV layout: bf16 [b][s][2304], K at col h*72, V at 1152 + h*72.
// Staging uses global_load_async_to_lds_b128 (rows are 144B = 9x16B).
// ---------------------------------------------------------------------------
__device__ __forceinline__ void stage_rows(const unsigned short* src, int c0,
                                           unsigned short* stage, int tid)
{
    unsigned base = lds_addr32(stage);
    for (int cid = tid; cid < 256 * 9; cid += 288) {
        int row = cid / 9;
        int seg = cid % 9;
        unsigned long long ga = (unsigned long long)(const void*)
            (src + (long)(c0 + row) * (2 * DD) + seg * 8);
        async_ld_b128(base + (unsigned)(row * 144 + seg * 16), ga);
    }
    wait_async0();
}

__global__ __launch_bounds__(288) void attn_kernel(const unsigned short* __restrict__ KV,
                                                   const float* __restrict__ qbuf,
                                                   float* __restrict__ ctx)
{
    const int h = blockIdx.x, b = blockIdx.y, tid = threadIdx.x;
    __shared__ float qs[TT][DHH];
    __shared__ float sc[TT][SS];
    __shared__ __align__(16) unsigned short stage[256 * DHH];
    __shared__ float red[TT][64];
    __shared__ float mrow[TT], srow[TT];

    if (tid < TT * DHH) {
        int t = tid / DHH, d = tid % DHH;
        qs[t][d] = qbuf[t * DD + h * DHH + d];
    }
    __syncthreads();

    const unsigned short* Kbase = KV + (long)b * SS * (2 * DD) + h * DHH;
    const unsigned short* Vbase = Kbase + DD;

    // scores
    for (int c0 = 0; c0 < SS; c0 += 256) {
        stage_rows(Kbase, c0, stage, tid);
        __syncthreads();
        if (tid < 256) {
            float a0 = 0.f, a1 = 0.f, a2 = 0.f, a3 = 0.f;
            for (int d = 0; d < DHH; ++d) {
                float kv = bf2f(stage[tid * DHH + d]);
                a0 += qs[0][d] * kv; a1 += qs[1][d] * kv;
                a2 += qs[2][d] * kv; a3 += qs[3][d] * kv;
            }
            sc[0][c0 + tid] = a0; sc[1][c0 + tid] = a1;
            sc[2][c0 + tid] = a2; sc[3][c0 + tid] = a3;
        }
        __syncthreads();
    }
    // softmax (max, exp, sum; 1/sum folded into ctx write)
    if (tid < 256) {
        int t = tid >> 6, j = tid & 63;
        float mx = -1e30f;
        for (int s = j; s < SS; s += 64) mx = fmaxf(mx, sc[t][s]);
        red[t][j] = mx;
    }
    __syncthreads();
    if (tid < TT) {
        float mx = red[tid][0];
        for (int j = 1; j < 64; ++j) mx = fmaxf(mx, red[tid][j]);
        mrow[tid] = mx;
    }
    __syncthreads();
    for (int idx = tid; idx < TT * SS; idx += 288) {
        int t = idx >> 10, s = idx & (SS - 1);
        sc[t][s] = __expf(sc[t][s] - mrow[t]);
    }
    __syncthreads();
    if (tid < 256) {
        int t = tid >> 6, j = tid & 63;
        float sm = 0.f;
        for (int s = j; s < SS; s += 64) sm += sc[t][s];
        red[t][j] = sm;
    }
    __syncthreads();
    if (tid < TT) {
        float sm = 0.f;
        for (int j = 0; j < 64; ++j) sm += red[tid][j];
        srow[tid] = sm;
    }
    __syncthreads();

    // ctx: thread owns (t, d)
    const int t = tid / DHH, d = tid % DHH;
    float accum = 0.f;
    for (int c0 = 0; c0 < SS; c0 += 256) {
        stage_rows(Vbase, c0, stage, tid);
        __syncthreads();
        for (int sl = 0; sl < 256; ++sl)
            accum += sc[t][c0 + sl] * bf2f(stage[sl * DHH + d]);
        __syncthreads();
    }
    ctx[((long)b * TT + t) * DD + h * DHH + d] = accum / srow[t];
}

// ---------------------------------------------------------------------------
// LayerNorm + gate logits + top-2 -> ln_out (bf16) and wfull (sparse weights)
// one block per token m = b*T + t
// ---------------------------------------------------------------------------
__global__ __launch_bounds__(256) void ln_gate_kernel(const float* __restrict__ resid,
                                                      const float* __restrict__ g,
                                                      const float* __restrict__ bta,
                                                      const float* __restrict__ gate_w,
                                                      const float* __restrict__ gate_b,
                                                      unsigned short* __restrict__ lnout,
                                                      float* __restrict__ wfull)
{
    const int m = blockIdx.x, tid = threadIdx.x, t = m & (TT - 1);
    __shared__ float x[DD];
    __shared__ float red[256];
    __shared__ float smu, srs;
    __shared__ float gl[EE];

    for (int d = tid; d < DD; d += 256) x[d] = resid[(long)m * DD + d];
    __syncthreads();

    float p = 0.f;
    for (int d = tid; d < DD; d += 256) p += x[d];
    red[tid] = p; __syncthreads();
    for (int s = 128; s > 0; s >>= 1) { if (tid < s) red[tid] += red[tid + s]; __syncthreads(); }
    if (tid == 0) smu = red[0] / (float)DD;
    __syncthreads();
    p = 0.f;
    for (int d = tid; d < DD; d += 256) { float v = x[d] - smu; p += v * v; }
    red[tid] = p; __syncthreads();
    for (int s = 128; s > 0; s >>= 1) { if (tid < s) red[tid] += red[tid + s]; __syncthreads(); }
    if (tid == 0) srs = rsqrtf(red[0] / (float)DD + 1e-6f);
    __syncthreads();
    for (int d = tid; d < DD; d += 256) {
        float y = (x[d] - smu) * srs * g[d] + bta[d];
        x[d] = y;
        lnout[(long)m * DD + d] = f2bf(y);
    }
    __syncthreads();

    // gate: logits[e] = sum_d x[d] * gate_w[t][d][e] + gate_b[t][e]
    {
        int e = tid & 7, j = tid >> 3;   // 32 partial groups per expert
        float s = 0.f;
        for (int d = j; d < DD; d += 32) s += x[d] * gate_w[((long)t * DD + d) * EE + e];
        red[tid] = s;
    }
    __syncthreads();
    if (tid < EE) {
        float s = gate_b[t * EE + tid];
        for (int j = 0; j < 32; ++j) s += red[j * EE + tid];
        gl[tid] = s;
        wfull[m * EE + tid] = 0.f;
    }
    __syncthreads();
    if (tid == 0) {
        int i1 = 0;
        for (int e = 1; e < EE; ++e) if (gl[e] > gl[i1]) i1 = e;
        int i2 = -1;
        for (int e = 0; e < EE; ++e)
            if (e != i1 && (i2 < 0 || gl[e] > gl[i2])) i2 = e;
        float e2 = __expf(gl[i2] - gl[i1]);
        float inv = 1.0f / (1.0f + e2);
        wfull[m * EE + i1] = inv;
        wfull[m * EE + i2] = e2 * inv;
    }
}

// ---------------------------------------------------------------------------
// out = residual + sum_e wfull[m][e] * fc2_b[e][:]   (fc2 GEMM atomically adds)
// ---------------------------------------------------------------------------
__global__ void init_out_kernel(const float* __restrict__ resid,
                                const float* __restrict__ wfull,
                                const float* __restrict__ fc2_b,
                                float* __restrict__ out)
{
    int idx = blockIdx.x * 256 + threadIdx.x;
    if (idx >= BB * TT * DD) return;
    int m = idx / DD, d = idx % DD;
    float s = resid[idx];
    for (int e = 0; e < EE; ++e) s += wfull[m * EE + e] * fc2_b[(long)e * DD + d];
    out[idx] = s;
}

// ---------------------------------------------------------------------------
extern "C" void kernel_launch(void* const* d_in, const int* in_sizes, int n_in,
                              void* d_out, int out_size, void* d_ws, size_t ws_size,
                              hipStream_t stream)
{
    const float* hs      = (const float*)d_in[0];
    const float* probe   = (const float*)d_in[1];
    const float* in_w    = (const float*)d_in[2];
    const float* in_b    = (const float*)d_in[3];
    const float* out_w   = (const float*)d_in[4];
    const float* out_b   = (const float*)d_in[5];
    const float* ln_g    = (const float*)d_in[6];
    const float* ln_b    = (const float*)d_in[7];
    const float* gate_w  = (const float*)d_in[8];
    const float* gate_b  = (const float*)d_in[9];
    const float* fc1_w   = (const float*)d_in[10];
    const float* fc1_b   = (const float*)d_in[11];
    const float* fc2_w   = (const float*)d_in[12];
    const float* fc2_b   = (const float*)d_in[13];
    float* out = (float*)d_out;

    char* w = (char*)d_ws;
    float*          qbuf  = (float*)w;          w += (size_t)TT * DD * 4;
    unsigned short* KV    = (unsigned short*)w; w += (size_t)BB * SS * 2 * DD * 2;
    float*          ctx   = (float*)w;          w += (size_t)BB * TT * DD * 4;
    float*          resid = (float*)w;          w += (size_t)BB * TT * DD * 4;
    unsigned short* lnout = (unsigned short*)w; w += (size_t)BB * TT * DD * 2;
    float*          wf    = (float*)w;          w += (size_t)BB * TT * EE * 4;
    unsigned short* hbuf  = (unsigned short*)w; w += (size_t)EE * BB * TT * FF_ * 2;

    // 1) q projection (tiny, batch-invariant), prescaled by 1/sqrt(DH)
    qproj_kernel<<<(TT * DD + 255) / 256, 256, 0, stream>>>(probe, in_w, in_b, qbuf);

    // 2) fused K/V projection: M=S, N=2D, K=D per batch -> bf16 KV
    wmma_gemm_kernel<float, 0><<<dim3(18, 8, BB), 256, 0, stream>>>(
        hs, (long)SS * DD,
        in_w + (long)DD * DD, 0L,
        in_b + DD, 0L,
        nullptr, KV, (long)SS * 2 * DD,
        nullptr, 0, 0L,
        SS, 2 * DD, DD, DD, DD, 2 * DD);

    // 3) attention -> ctx (B,T,D) fp32
    attn_kernel<<<dim3(NHH, BB), 288, 0, stream>>>(KV, qbuf, ctx);

    // 4) out_proj: (256 x 1152) = ctx @ Wout^T + b  -> residual
    wmma_gemm_kernel<float, 1><<<dim3(9, 2, 1), 256, 0, stream>>>(
        ctx, 0L, out_w, 0L, out_b, 0L,
        resid, nullptr, 0L, nullptr, 0, 0L,
        BB * TT, DD, DD, DD, DD, DD);

    // 5) LayerNorm + gate + top-2
    ln_gate_kernel<<<BB * TT, 256, 0, stream>>>(resid, ln_g, ln_b, gate_w, gate_b, lnout, wf);

    // 6) out = residual + sum_e wfull*fc2_b
    init_out_kernel<<<(BB * TT * DD + 255) / 256, 256, 0, stream>>>(resid, wf, fc2_b, out);

    // 7) fc1 per expert: h = wfull * gelu(ln @ fc1_w^T + b1) -> bf16 (async A path)
    wmma_gemm_kernel<unsigned short, 2><<<dim3(34, 2, EE), 256, 0, stream>>>(
        lnout, 0L,
        fc1_w, (long)FF_ * DD,
        fc1_b, (long)FF_,
        nullptr, hbuf, (long)BB * TT * FF_,
        wf, EE, 1L,
        BB * TT, FF_, DD, DD, DD, FF_);

    // 8) fc2 per expert: out += h @ fc2_w^T  (atomic f32 accumulate, async A path)
    wmma_gemm_kernel<unsigned short, 3><<<dim3(9, 2, EE), 256, 0, stream>>>(
        hbuf, (long)BB * TT * FF_,
        fc2_w, (long)DD * FF_,
        nullptr, 0L,
        out, nullptr, 0L,
        nullptr, 0, 0L,
        BB * TT, DD, FF_, FF_, FF_, DD);
}
